// EquivariantUNet_56169582297229
// MI455X (gfx1250) — matrix operand order repaired and checked by reading, not compiled, not run on hardware
//
#include <hip/hip_runtime.h>
#include <math.h>

typedef __attribute__((ext_vector_type(16))) __bf16 v16bf;
typedef __attribute__((ext_vector_type(8)))  float  v8f;

#define D_      256
#define NNODES  8192
#define NEDGES  262144
#define CHUNK   65536
#define ALEN    512
#define NHEADS  8
#define HD      32

// ---------- bf16 <-> f32 bit helpers ----------

__device__ __forceinline__ unsigned short f2bf_bits(float f) {
  union { float f; unsigned u; } x; x.f = f;
  unsigned r = x.u + 0x7FFFu + ((x.u >> 16) & 1u);   // round-to-nearest-even
  return (unsigned short)(r >> 16);
}
__device__ __forceinline__ float bf2f(unsigned short s) {
  union { unsigned u; float f; } x; x.u = ((unsigned)s) << 16;
  return x.f;
}
__device__ __forceinline__ float silu_f(float x) {
  return x / (1.0f + __expf(-x));
}

// ---------- CDNA5 async global->LDS copy (ASYNCcnt path) ----------

template <typename T>
__device__ __forceinline__ unsigned lds_byte_addr(T* p) {
  return (unsigned)(unsigned long long)(__attribute__((address_space(3))) T*)p;
}
__device__ __forceinline__ void async_copy_b128(unsigned lds_off, const void* g) {
  unsigned long long ga = (unsigned long long)g;
  asm volatile("global_load_async_to_lds_b128 %0, %1, off"
               :: "v"(lds_off), "v"(ga) : "memory");
}
__device__ __forceinline__ void wait_async0() {
  asm volatile("s_wait_asynccnt 0" ::: "memory");
}
__device__ __forceinline__ void wait_async3() {   // allow next stage's 3 copies in flight
  asm volatile("s_wait_asynccnt 3" ::: "memory");
}

// ---------- WMMA fragment loaders (LDS bf16 tiles) ----------
// A: [m][k] row-major, row stride 32 bf16. B: [n][k] row-major, stride 32 bf16.

__device__ __forceinline__ v16bf load_a_frag(const unsigned short* sA, int m0, int lane) {
  const int half = lane >> 4;
  const unsigned short* rp = sA + (m0 + (lane & 15)) * 32;
  v16bf f;
#pragma unroll
  for (int j = 0; j < 8; ++j) {
    int e = 2 * j;
    int k = half * 16 + e;
    unsigned u = *reinterpret_cast<const unsigned*>(rp + k);
    union { unsigned u; __bf16 b[2]; } c; c.u = u;
    f[e] = c.b[0]; f[e + 1] = c.b[1];
  }
  return f;
}
__device__ __forceinline__ v16bf load_b_frag(const unsigned short* sBt, int n0, int lane) {
  const int half = lane >> 4;
  const unsigned short* rp = sBt + (n0 + (lane & 15)) * 32 + half * 16;
  v16bf f;
#pragma unroll
  for (int j = 0; j < 8; ++j) {
    unsigned u = *reinterpret_cast<const unsigned*>(rp + 2 * j);
    union { unsigned u; __bf16 b[2]; } c; c.u = u;
    f[2 * j] = c.b[0]; f[2 * j + 1] = c.b[1];
  }
  return f;
}

// ---------- GEMM: C = act(A_bf16[MxK] @ Wt_bf16[NxK]^T + bias) ----------
// 128x64 block tile, 8 waves in 4x2 grid, each wave a 32x32 micro-tile
// (2x2 WMMAs with fragment reuse). K stepped by 32 with double-buffered
// async-to-LDS copies (3 copy instructions per wave per stage).

template <int ACT, int OUTF32>
__global__ __launch_bounds__(256) void gemm_bf16(
    const unsigned short* __restrict__ A, const unsigned short* __restrict__ Wt,
    const float* __restrict__ bias, void* __restrict__ Cv,
    int M, int N, int K, int lda, int ldb) {
  __shared__ __align__(16) unsigned short sA[2][128 * 32];
  __shared__ __align__(16) unsigned short sB[2][64 * 32];
  const int tid  = threadIdx.x;
  const int lane = tid & 31;
  const int wave = tid >> 5;
  const int wn = wave & 1;         // 32-col group
  const int wm = wave >> 1;        // 32-row group (0..3)
  const int bm = blockIdx.y * 128;
  const int bn = blockIdx.x * 64;
  const int cr = tid >> 2, seg = tid & 3;       // copy row / 16B segment

  auto issue = [&](int k0, int buf) {
    unsigned laA = lds_byte_addr(&sA[buf][0]);
    unsigned laB = lds_byte_addr(&sB[buf][0]);
    async_copy_b128(laA + cr * 64 + seg * 16,
                    A + (size_t)(bm + cr) * lda + k0 + seg * 8);
    async_copy_b128(laA + (cr + 64) * 64 + seg * 16,
                    A + (size_t)(bm + cr + 64) * lda + k0 + seg * 8);
    async_copy_b128(laB + cr * 64 + seg * 16,
                    Wt + (size_t)(bn + cr) * ldb + k0 + seg * 8);
  };

  v8f acc00 = {}, acc01 = {}, acc10 = {}, acc11 = {};
  const int nsteps = K >> 5;       // K is always a multiple of 32 here

  issue(0, 0);
  for (int s = 0; s < nsteps; ++s) {
    if (s + 1 < nsteps) { issue((s + 1) * 32, (s + 1) & 1); wait_async3(); }
    else                { wait_async0(); }
    __syncthreads();
    const unsigned short* cA = &sA[s & 1][0];
    const unsigned short* cB = &sB[s & 1][0];
    v16bf fa0 = load_a_frag(cA, wm * 32, lane);
    v16bf fa1 = load_a_frag(cA, wm * 32 + 16, lane);
    v16bf fb0 = load_b_frag(cB, wn * 32, lane);
    v16bf fb1 = load_b_frag(cB, wn * 32 + 16, lane);
    acc00 = __builtin_amdgcn_wmma_f32_16x16x32_bf16(false, fa0, false, fb0, (short)0, acc00, false, false);
    acc01 = __builtin_amdgcn_wmma_f32_16x16x32_bf16(false, fa0, false, fb1, (short)0, acc01, false, false);
    acc10 = __builtin_amdgcn_wmma_f32_16x16x32_bf16(false, fa1, false, fb0, (short)0, acc10, false, false);
    acc11 = __builtin_amdgcn_wmma_f32_16x16x32_bf16(false, fa1, false, fb1, (short)0, acc11, false, false);
    __syncthreads();
  }

  const int half = lane >> 4;
  const int n0 = bn + wn * 32 + (lane & 15);
  const int n1 = n0 + 16;
  const float bv0 = bias[n0];
  const float bv1 = bias[n1];
#pragma unroll
  for (int r = 0; r < 8; ++r) {
    int m0 = bm + wm * 32 + r + 8 * half;
    float v00 = acc00[r] + bv0, v01 = acc01[r] + bv1;
    float v10 = acc10[r] + bv0, v11 = acc11[r] + bv1;
    if (ACT == 1) { v00 = silu_f(v00); v01 = silu_f(v01); v10 = silu_f(v10); v11 = silu_f(v11); }
    if (OUTF32) {
      float* C = (float*)Cv;
      C[(size_t)m0 * N + n0]        = v00;
      C[(size_t)m0 * N + n1]        = v01;
      C[(size_t)(m0 + 16) * N + n0] = v10;
      C[(size_t)(m0 + 16) * N + n1] = v11;
    } else {
      unsigned short* C = (unsigned short*)Cv;
      C[(size_t)m0 * N + n0]        = f2bf_bits(v00);
      C[(size_t)m0 * N + n1]        = f2bf_bits(v01);
      C[(size_t)(m0 + 16) * N + n0] = f2bf_bits(v10);
      C[(size_t)(m0 + 16) * N + n1] = f2bf_bits(v11);
    }
  }
}

// ---------- weight pre-transpose+convert: Wt[n][k] bf16 (K padded to KS) ----------

__global__ void wconv_kernel(const float* __restrict__ W, unsigned short* __restrict__ Wt,
                             int K, int N, int KS) {
  int i = blockIdx.x * 256 + threadIdx.x;
  if (i >= N * KS) return;
  int n = i / KS, k = i - n * KS;
  float v = (k < K) ? W[(size_t)k * N + n] : 0.0f;
  Wt[i] = f2bf_bits(v);
}

// ---------- f32 -> bf16 convert ----------

__global__ void f32_to_bf16_kernel(const float* __restrict__ a, unsigned short* __restrict__ o, int n) {
  int i = blockIdx.x * 256 + threadIdx.x;
  if (i < n) o[i] = f2bf_bits(a[i]);
}

// ---------- edge features -> ef[CHUNK x 32] bf16 (cols 12..31 zero) ----------

__global__ __launch_bounds__(256) void edge_feat_kernel(
    const float* __restrict__ pos, const float* __restrict__ edge_attr,
    const int* __restrict__ ei, unsigned short* __restrict__ ef, int e0) {
  int le = blockIdx.x * 256 + threadIdx.x;
  int e = e0 + le;
  int r = ei[e];
  int c = ei[NEDGES + e];
  float dx = pos[r * 3 + 0] - pos[c * 3 + 0];
  float dy = pos[r * 3 + 1] - pos[c * 3 + 1];
  float dz = pos[r * 3 + 2] - pos[c * 3 + 2];
  float len = sqrtf(dx * dx + dy * dy + dz * dz);
  float inv = 1.0f / (len + 1e-8f);
  float x = dx * inv, y = dy * inv, z = dz * inv;
  const float s3 = 1.7320508075688772f, s5 = 2.2360679774997896f, s15 = 3.8729833462074170f;
  unsigned short* o = ef + (size_t)le * 32;
  o[0] = f2bf_bits(1.0f);
  o[1] = f2bf_bits(s3 * x); o[2] = f2bf_bits(s3 * y); o[3] = f2bf_bits(s3 * z);
  o[4] = f2bf_bits(s15 * x * y); o[5] = f2bf_bits(s15 * y * z);
  o[6] = f2bf_bits(0.5f * s5 * (3.0f * z * z - 1.0f));
  o[7] = f2bf_bits(s15 * x * z);
  o[8] = f2bf_bits(0.5f * s15 * (x * x - y * y));
  o[9]  = f2bf_bits(edge_attr[(size_t)e * 3 + 0]);
  o[10] = f2bf_bits(edge_attr[(size_t)e * 3 + 1]);
  o[11] = f2bf_bits(edge_attr[(size_t)e * 3 + 2]);
#pragma unroll
  for (int j = 12; j < 32; ++j) o[j] = 0;
}

// ---------- msg = xt[row] * em ; scatter-add fp32 ----------

__global__ __launch_bounds__(256) void scatter_kernel(
    const unsigned short* __restrict__ xt, const unsigned short* __restrict__ em,
    const int* __restrict__ ei, float* __restrict__ agg,
    float* __restrict__ cnt, int e0) {
  const int le = blockIdx.x;
  const int e = e0 + le;
  const int r = ei[e];
  const int c = ei[NEDGES + e];
  const int d = threadIdx.x;
  __builtin_prefetch(em + ((size_t)le + 1) * D_ + d, 0, 1);
  float m = bf2f(xt[(size_t)r * D_ + d]) * bf2f(em[(size_t)le * D_ + d]);
  atomicAdd(agg + (size_t)c * D_ + d, m);
  if (d == 0) atomicAdd(cnt + c, 1.0f);
}

// ---------- misc elementwise ----------

__global__ void fill_kernel(float* __restrict__ p, float v, int n) {
  int i = blockIdx.x * 256 + threadIdx.x;
  if (i < n) p[i] = v;
}

__global__ void concat_kernel(const float* __restrict__ agg, const float* __restrict__ cnt,
                              const unsigned short* __restrict__ h,
                              unsigned short* __restrict__ cat) {
  int i = blockIdx.x * 256 + threadIdx.x;       // over NNODES*512
  int nrow = i >> 9; int j = i & 511;
  float v;
  if (j < D_) v = agg[(size_t)nrow * D_ + j] / fmaxf(cnt[nrow], 1.0f);
  else        v = bf2f(h[(size_t)nrow * D_ + (j - D_)]);
  cat[i] = f2bf_bits(v);
}

// ---------- batchnorm (bf16 tensor, fp32 stats) ----------

__global__ __launch_bounds__(256) void bn_stats_kernel(const unsigned short* __restrict__ h,
                                                       float* __restrict__ stats) {
  __shared__ float s1[256], s2[256];
  int col = blockIdx.x;
  float sum = 0.0f, sq = 0.0f;
  for (int r = threadIdx.x; r < NNODES; r += 256) {
    float v = bf2f(h[(size_t)r * D_ + col]);
    sum += v; sq += v * v;
  }
  s1[threadIdx.x] = sum; s2[threadIdx.x] = sq;
  __syncthreads();
  for (int off = 128; off > 0; off >>= 1) {
    if (threadIdx.x < off) {
      s1[threadIdx.x] += s1[threadIdx.x + off];
      s2[threadIdx.x] += s2[threadIdx.x + off];
    }
    __syncthreads();
  }
  if (threadIdx.x == 0) {
    float mean = s1[0] / (float)NNODES;
    float var  = s2[0] / (float)NNODES - mean * mean;
    stats[col]      = mean;
    stats[D_ + col] = rsqrtf(var + 1e-5f);
  }
}

template <int ACT>
__global__ void bn_apply_kernel(unsigned short* __restrict__ h, const float* __restrict__ stats,
                                const float* __restrict__ g, const float* __restrict__ b) {
  int i = blockIdx.x * 256 + threadIdx.x;
  if (i >= NNODES * D_) return;
  int col = i & (D_ - 1);
  float v = (bf2f(h[i]) - stats[col]) * stats[D_ + col] * g[col] + b[col];
  if (ACT == 1) v = silu_f(v);
  h[i] = f2bf_bits(v);
}

// ---------- attention ----------

__global__ __launch_bounds__(256) void attn_scores_kernel(
    const unsigned short* __restrict__ q, const unsigned short* __restrict__ k,
    float* __restrict__ scores) {
  __shared__ __align__(16) unsigned short sQ[64 * 32];
  __shared__ __align__(16) unsigned short sK[64 * 32];
  const int tid = threadIdx.x, lane = tid & 31, wave = tid >> 5;
  const int wn = wave & 3, wm = wave >> 2;
  const int bh = blockIdx.z, b = bh >> 3, hh = bh & 7;
  const int bm = blockIdx.y * 64, bn = blockIdx.x * 64;
  const unsigned laQ = lds_byte_addr(sQ);
  const unsigned laK = lds_byte_addr(sK);
  const int cr = tid >> 2, seg = tid & 3;
  async_copy_b128(laQ + cr * 64 + seg * 16,
                  q + (size_t)(b * ALEN + bm + cr) * D_ + hh * HD + seg * 8);
  async_copy_b128(laK + cr * 64 + seg * 16,
                  k + (size_t)(b * ALEN + bn + cr) * D_ + hh * HD + seg * 8);
  wait_async0();
  __syncthreads();
  v16bf fb  = load_b_frag(sK, wn * 16, lane);
  v16bf fa0 = load_a_frag(sQ, wm * 32, lane);
  v16bf fa1 = load_a_frag(sQ, wm * 32 + 16, lane);
  v8f acc0 = {}, acc1 = {};
  acc0 = __builtin_amdgcn_wmma_f32_16x16x32_bf16(false, fa0, false, fb, (short)0, acc0, false, false);
  acc1 = __builtin_amdgcn_wmma_f32_16x16x32_bf16(false, fa1, false, fb, (short)0, acc1, false, false);
  const float sc = 0.17677669529663687f;   // 1/sqrt(32)
  const int half = lane >> 4;
  const int n = bn + wn * 16 + (lane & 15);
  float* srow = scores + (size_t)bh * ALEN * ALEN;
#pragma unroll
  for (int r = 0; r < 8; ++r) {
    int m0 = bm + wm * 32 + r + 8 * half;
    srow[(size_t)m0 * ALEN + n]        = acc0[r] * sc;
    srow[(size_t)(m0 + 16) * ALEN + n] = acc1[r] * sc;
  }
}

__global__ __launch_bounds__(128) void softmax_kernel(const float* __restrict__ s,
                                                      unsigned short* __restrict__ p) {
  __shared__ float red[128];
  const float* row = s + (size_t)blockIdx.x * ALEN;
  unsigned short* prow = p + (size_t)blockIdx.x * ALEN;
  int t = threadIdx.x;
  float v[4];
  float mx = -1e30f;
#pragma unroll
  for (int i = 0; i < 4; ++i) { v[i] = row[t + 128 * i]; mx = fmaxf(mx, v[i]); }
  red[t] = mx; __syncthreads();
  for (int off = 64; off > 0; off >>= 1) {
    if (t < off) red[t] = fmaxf(red[t], red[t + off]);
    __syncthreads();
  }
  mx = red[0]; __syncthreads();
  float sum = 0.0f;
#pragma unroll
  for (int i = 0; i < 4; ++i) { v[i] = __expf(v[i] - mx); sum += v[i]; }
  red[t] = sum; __syncthreads();
  for (int off = 64; off > 0; off >>= 1) {
    if (t < off) red[t] += red[t + off];
    __syncthreads();
  }
  float inv = 1.0f / red[0];
#pragma unroll
  for (int i = 0; i < 4; ++i) prow[t + 128 * i] = f2bf_bits(v[i] * inv);
}

// o[b, m, hh*32+n] = sum_k probs[bh][m][k] * v_bh[k][n]

__global__ __launch_bounds__(256) void attn_av_kernel(
    const unsigned short* __restrict__ pr, const unsigned short* __restrict__ v,
    unsigned short* __restrict__ o) {
  __shared__ __align__(16) unsigned short sS[64 * 32];
  __shared__ __align__(16) unsigned short sV[32 * 32];
  const int tid = threadIdx.x, lane = tid & 31, wave = tid >> 5;
  const int wn = wave & 1, wm = wave >> 1;       // 4x2 wave grid -> 64x32 tile
  const int bh = blockIdx.y, b = bh >> 3, hh = bh & 7;
  const int bm = blockIdx.x * 64;
  const unsigned short* prow = pr + (size_t)bh * ALEN * ALEN;
  const unsigned laS = lds_byte_addr(sS);
  const int cr = tid >> 2, seg = tid & 3;
  v8f acc = {};
  for (int k0 = 0; k0 < ALEN; k0 += 32) {
    async_copy_b128(laS + cr * 64 + seg * 16,
                    prow + (size_t)(bm + cr) * ALEN + k0 + seg * 8);
    // V tile transposed via VGPR staging: sV[n][kk] = v_bh[k0+kk][n]
#pragma unroll
    for (int i = 0; i < 4; ++i) {
      int l = tid + 256 * i;
      int kk = l >> 5, n = l & 31;
      sV[n * 32 + kk] = v[(size_t)(b * ALEN + k0 + kk) * D_ + hh * HD + n];
    }
    wait_async0();
    __syncthreads();
    v16bf fb = load_b_frag(sV, wn * 16, lane);
    v16bf fa = load_a_frag(sS, wm * 16, lane);
    acc = __builtin_amdgcn_wmma_f32_16x16x32_bf16(false, fa, false, fb, (short)0, acc, false, false);
    __syncthreads();
  }
  const int half = lane >> 4;
  const int n = wn * 16 + (lane & 15);
#pragma unroll
  for (int r = 0; r < 8; ++r) {
    int m0 = bm + wm * 16 + r + 8 * half;
    o[(size_t)(b * ALEN + m0) * D_ + hh * HD + n] = f2bf_bits(acc[r]);
  }
}

// ---------- host orchestration ----------

extern "C" void kernel_launch(void* const* d_in, const int* in_sizes, int n_in,
                              void* d_out, int out_size, void* d_ws, size_t ws_size,
                              hipStream_t stream) {
  const float* x         = (const float*)d_in[0];
  const float* edge_attr = (const float*)d_in[1];
  const float* pos       = (const float*)d_in[2];
  const float* P[40];
  for (int i = 0; i < 40; ++i) P[i] = (const float*)d_in[3 + i];
  const int* ei = (const int*)d_in[43];

  const size_t ND = (size_t)NNODES * D_;
  char* base = (char*)d_ws;
  size_t ofs = 0;
  auto carve = [&](size_t bytes) { void* p = base + ofs; ofs += (bytes + 255) & ~(size_t)255; return p; };

  // region0: edge phase (eh+em bf16) later reused by attention (scores f32 + probs bf16)
  char* region0 = (char*)carve((size_t)128 * 512 * 512 * 4 + (size_t)128 * 512 * 512 * 2);
  unsigned short* eh = (unsigned short*)region0;                         // CHUNK*D bf16
  unsigned short* em = eh + (size_t)CHUNK * D_;                          // CHUNK*D bf16
  float*          scores = (float*)region0;                              // 128*512*512 f32
  unsigned short* pb = (unsigned short*)(region0 + (size_t)128 * 512 * 512 * 4);

  unsigned short* efb  = (unsigned short*)carve((size_t)CHUNK * 32 * 2);
  unsigned short* tmp1 = (unsigned short*)carve(ND * 2);
  unsigned short* xt   = (unsigned short*)carve(ND * 2);
  float*          agg  = (float*)carve(ND * 4);
  float*          cnt  = (float*)carve((size_t)NNODES * 4);
  unsigned short* cat  = (unsigned short*)carve(2 * ND * 2);
  unsigned short* hbuf = (unsigned short*)carve(ND * 2);
  unsigned short* qb   = (unsigned short*)carve(ND * 2);
  unsigned short* kb   = (unsigned short*)carve(ND * 2);
  unsigned short* vb   = (unsigned short*)carve(ND * 2);
  unsigned short* ob   = (unsigned short*)carve(ND * 2);
  unsigned short* xbf  = (unsigned short*)carve(ND * 2);
  float*          stats = (float*)carve((size_t)2 * D_ * 4);

  // pre-transposed bf16 weights: [6 per conv layer x2, then q,k,v,out]
  const int wk[16]  = {256,256, 12,256, 512,256,  256,256, 12,256, 512,256,  256,256,256,256};
  const int wks[16] = {256,256, 32,256, 512,256,  256,256, 32,256, 512,256,  256,256,256,256};
  const int wpi[16] = {3,5, 7,9, 11,13,  17,19, 21,23, 25,27,  35,37,39,41}; // d_in index of weight
  unsigned short* wt[16];
  for (int i = 0; i < 16; ++i) {
    int elems = 256 * wks[i];
    wt[i] = (unsigned short*)carve((size_t)elems * 2);
    wconv_kernel<<<(elems + 255) / 256, 256, 0, stream>>>(
        (const float*)d_in[wpi[i]], wt[i], wk[i], 256, wks[i]);
  }

  auto gemm = [&](const unsigned short* A, int widx, void* C,
                  int M, int K, int lda, int act, int outf32) {
    dim3 g(256 / 64, M / 128);
    const unsigned short* W = wt[widx];
    const float* bias = (const float*)d_in[wpi[widx] + 1];
    int ldb = wks[widx];
    if (act)         gemm_bf16<1, 0><<<g, 256, 0, stream>>>(A, W, bias, C, M, 256, K, lda, ldb);
    else if (outf32) gemm_bf16<0, 1><<<g, 256, 0, stream>>>(A, W, bias, C, M, 256, K, lda, ldb);
    else             gemm_bf16<0, 0><<<g, 256, 0, stream>>>(A, W, bias, C, M, 256, K, lda, ldb);
  };

  auto conv = [&](int layer, const unsigned short* h_in, unsigned short* h_out) {
    int w0 = layer * 6;
    const float* bn_g = P[layer * 14 + 12];
    const float* bn_b = P[layer * 14 + 13];
    // node MLP
    gemm(h_in, w0 + 0, tmp1, NNODES, 256, 256, 1, 0);
    gemm(tmp1, w0 + 1, xt,   NNODES, 256, 256, 0, 0);
    // aggregation buffers
    fill_kernel<<<(int)(ND / 256), 256, 0, stream>>>(agg, 0.0f, (int)ND);
    fill_kernel<<<NNODES / 256, 256, 0, stream>>>(cnt, 0.0f, NNODES);
    // edge MLP + scatter, chunked
    for (int c = 0; c < NEDGES / CHUNK; ++c) {
      int e0 = c * CHUNK;
      edge_feat_kernel<<<CHUNK / 256, 256, 0, stream>>>(pos, edge_attr, ei, efb, e0);
      gemm(efb, w0 + 2, eh, CHUNK, 32, 32, 1, 0);
      gemm(eh,  w0 + 3, em, CHUNK, 256, 256, 0, 0);
      scatter_kernel<<<CHUNK, 256, 0, stream>>>(xt, em, ei, agg, cnt, e0);
    }
    // out MLP on [agg/cnt, h_in]
    concat_kernel<<<(int)(2 * ND / 256), 256, 0, stream>>>(agg, cnt, h_in, cat);
    gemm(cat,  w0 + 4, tmp1,  NNODES, 512, 512, 1, 0);
    gemm(tmp1, w0 + 5, h_out, NNODES, 256, 256, 0, 0);
    // conv batchnorm
    bn_stats_kernel<<<D_, 256, 0, stream>>>(h_out, stats);
    bn_apply_kernel<0><<<(int)(ND / 256), 256, 0, stream>>>(h_out, stats, bn_g, bn_b);
  };

  // bf16 copy of x for conv1 input
  f32_to_bf16_kernel<<<(int)(ND / 256), 256, 0, stream>>>(x, xbf, (int)ND);

  // conv1 -> bn(norm1) -> silu
  conv(0, xbf, hbuf);
  bn_stats_kernel<<<D_, 256, 0, stream>>>(hbuf, stats);
  bn_apply_kernel<1><<<(int)(ND / 256), 256, 0, stream>>>(hbuf, stats, P[28], P[29]);
  // conv2 -> bn(norm2) -> silu
  conv(1, hbuf, hbuf);
  bn_stats_kernel<<<D_, 256, 0, stream>>>(hbuf, stats);
  bn_apply_kernel<1><<<(int)(ND / 256), 256, 0, stream>>>(hbuf, stats, P[30], P[31]);

  // attention
  gemm(hbuf, 12, qb, NNODES, 256, 256, 0, 0);
  gemm(hbuf, 13, kb, NNODES, 256, 256, 0, 0);
  gemm(hbuf, 14, vb, NNODES, 256, 256, 0, 0);
  attn_scores_kernel<<<dim3(ALEN / 64, ALEN / 64, 16 * NHEADS), 256, 0, stream>>>(qb, kb, scores);
  softmax_kernel<<<16 * NHEADS * ALEN, 128, 0, stream>>>(scores, pb);
  attn_av_kernel<<<dim3(ALEN / 64, 16 * NHEADS), 256, 0, stream>>>(pb, vb, ob);
  gemm(ob, 15, (float*)d_out, NNODES, 256, 256, 0, 1);
}
